// Attention_32787780337677
// MI455X (gfx1250) — matrix operand order repaired and checked by reading, not compiled, not run on hardware
//
#include <hip/hip_runtime.h>

// ---------------- types ----------------
typedef __bf16 v16bf __attribute__((ext_vector_type(16)));
typedef __bf16 v8bf  __attribute__((ext_vector_type(8)));
typedef __bf16 v4bf  __attribute__((ext_vector_type(4)));
typedef float  v8f   __attribute__((ext_vector_type(8)));

union BF { v16bf v; v8bf h[2]; };

// ---------------- constants ----------------
constexpr int Bdim  = 2;
constexpr int Sdim  = 2048;
constexpr int Ddim  = 1024;
constexpr int Hdim  = 16;
constexpr int HDdim = 64;

// ---------------- helpers ----------------
__device__ inline __bf16 f2bf(float f) {
  unsigned u = __builtin_bit_cast(unsigned, f);
  unsigned r = u + 0x7FFFu + ((u >> 16) & 1u);   // round-to-nearest-even
  unsigned short hsd = (unsigned short)(r >> 16);
  return __builtin_bit_cast(__bf16, hsd);
}

__device__ inline v8bf ld8(const __bf16* p) { return *(const v8bf*)p; }

__device__ inline v8f wmma_bf16(v16bf a, v16bf b, v8f c) {
  return __builtin_amdgcn_wmma_f32_16x16x32_bf16(
      /*neg_a=*/false, a, /*neg_b=*/false, b,
      /*c_mod=*/(short)0, c, /*reuse_a=*/false, /*reuse_b=*/false);
}

// ---------------- kernel 1: x fp32 -> bf16 ----------------
__global__ void k_conv_x(const float4* __restrict__ in, v4bf* __restrict__ out, int n4) {
  int i = blockIdx.x * blockDim.x + threadIdx.x;
  if (i < n4) {
    float4 f = in[i];
    v4bf o;
    o[0] = f2bf(f.x); o[1] = f2bf(f.y); o[2] = f2bf(f.z); o[3] = f2bf(f.w);
    out[i] = o;
  }
}

// ---------------- kernel 2: transpose weights fp32[k][n] -> bf16[n][k] ----------------
__global__ void k_wt(const float* __restrict__ w0, const float* __restrict__ w1,
                     const float* __restrict__ w2, const float* __restrict__ w3,
                     __bf16* __restrict__ t0, __bf16* __restrict__ t1,
                     __bf16* __restrict__ t2, __bf16* __restrict__ t3) {
  __shared__ float tile[16][17];
  const float* src; __bf16* dst;
  switch (blockIdx.z) {
    case 0:  src = w0; dst = t0; break;
    case 1:  src = w1; dst = t1; break;
    case 2:  src = w2; dst = t2; break;
    default: src = w3; dst = t3; break;
  }
  int k0 = blockIdx.x * 16, n0 = blockIdx.y * 16;
  tile[threadIdx.y][threadIdx.x] = src[(size_t)(k0 + threadIdx.y) * Ddim + (n0 + threadIdx.x)];
  __syncthreads();
  dst[(size_t)(n0 + threadIdx.y) * Ddim + (k0 + threadIdx.x)] = f2bf(tile[threadIdx.x][threadIdx.y]);
}

// ---------------- shared 32x64 MMA core with double-buffered K loop ----------------
__device__ inline void mma_32x64(const __bf16* __restrict__ Abase,
                                 const __bf16* __restrict__ Wt,
                                 int mb, int nb, int ln, int hs, v8f acc[2][4]) {
  const __bf16* ar0 = Abase + (size_t)(mb + ln) * Ddim + hs * 8;
  const __bf16* ar1 = Abase + (size_t)(mb + 16 + ln) * Ddim + hs * 8;
  const __bf16* brj[4];
#pragma unroll
  for (int j = 0; j < 4; ++j)
    brj[j] = Wt + (size_t)(nb + j * 16 + ln) * Ddim + hs * 8;

  BF a0[2], b0[4], a1[2], b1[4];

  auto loadS = [&](BF aS[2], BF bS[4], int k0) {
    aS[0].h[0] = ld8(ar0 + k0); aS[0].h[1] = ld8(ar0 + k0 + 16);
    aS[1].h[0] = ld8(ar1 + k0); aS[1].h[1] = ld8(ar1 + k0 + 16);
#pragma unroll
    for (int j = 0; j < 4; ++j) {
      bS[j].h[0] = ld8(brj[j] + k0); bS[j].h[1] = ld8(brj[j] + k0 + 16);
    }
  };
  auto mmaS = [&](BF aS[2], BF bS[4]) {
#pragma unroll
    for (int i = 0; i < 2; ++i)
#pragma unroll
      for (int j = 0; j < 4; ++j)
        acc[i][j] = wmma_bf16(aS[i].v, bS[j].v, acc[i][j]);
  };

  loadS(a0, b0, 0);
#pragma unroll 1
  for (int k0 = 0; k0 < Ddim; k0 += 64) {
    loadS(a1, b1, k0 + 32);           // prefetch stage 1
    mmaS(a0, b0);                     // compute stage 0 while stage 1 in flight
    if (k0 + 64 < Ddim) loadS(a0, b0, k0 + 64);  // prefetch next stage 0
    mmaS(a1, b1);
  }
}

// ---------------- kernel 3: QKV projection (one wave = 32x64 tile) ----------------
__global__ __launch_bounds__(32) void k_qkv(
    const __bf16* __restrict__ xb,
    const __bf16* __restrict__ WqT, const __bf16* __restrict__ WkT, const __bf16* __restrict__ WvT,
    const float* __restrict__ bq, const float* __restrict__ bk, const float* __restrict__ bv,
    __bf16* __restrict__ Qb, __bf16* __restrict__ Kb, __bf16* __restrict__ Vtb) {
  int lane = threadIdx.x & 31, hs = lane >> 4, ln = lane & 15;
  int mb = blockIdx.x * 32, nb = blockIdx.y * 64, w = blockIdx.z;
  const __bf16* Wt   = (w == 0) ? WqT : ((w == 1) ? WkT : WvT);
  const float*  bias = (w == 0) ? bq  : ((w == 1) ? bk  : bv);

  v8f acc[2][4] = {};
  mma_32x64(xb, Wt, mb, nb, ln, hs, acc);

#pragma unroll
  for (int j = 0; j < 4; ++j) {
    int n = nb + j * 16 + ln;
    float bvl = bias[n];
    int h = n >> 6, e = n & 63;
#pragma unroll
    for (int i = 0; i < 2; ++i)
#pragma unroll
      for (int r = 0; r < 8; ++r) {
        int m  = mb + i * 16 + r + 8 * hs;
        int bb = m >> 11, s = m & 2047;
        float val = acc[i][j][r] + bvl;
        if (w < 2) {  // Q, K: [B,H,S,HD]
          __bf16* dst = (w == 0) ? Qb : Kb;
          dst[(((size_t)bb * Hdim + h) * Sdim + s) * HDdim + e] = f2bf(val);
        } else {      // V transposed: [B,H,HD,S]
          Vtb[(((size_t)bb * Hdim + h) * HDdim + e) * Sdim + s] = f2bf(val);
        }
      }
  }
}

// ---------------- kernel 4: causal flash attention (one wave = 32 queries) ----------------
__global__ __launch_bounds__(32) void k_attn(
    const __bf16* __restrict__ Qb, const __bf16* __restrict__ Kb,
    const __bf16* __restrict__ Vtb, __bf16* __restrict__ ctx) {
  __shared__ __bf16 lds_p[2 * 16 * 32];
  int lane = threadIdx.x & 31, hs = lane >> 4, ln = lane & 15;
  int qb32 = blockIdx.x * 32, h = blockIdx.y, b = blockIdx.z;
  size_t bh = (size_t)b * Hdim + h;
  const __bf16* Qh = Qb  + bh * Sdim * HDdim;
  const __bf16* Kh = Kb  + bh * Sdim * HDdim;
  const __bf16* Vh = Vtb + bh * HDdim * Sdim;

  // Q A-fragments: two query tiles x two K=32 chunks of HD=64
  BF aq[2][2];
#pragma unroll
  for (int t = 0; t < 2; ++t)
#pragma unroll
    for (int c = 0; c < 2; ++c) {
      const __bf16* qr = Qh + (size_t)(qb32 + t * 16 + ln) * HDdim + c * 32 + hs * 8;
      aq[t][c].h[0] = ld8(qr);
      aq[t][c].h[1] = ld8(qr + 16);
    }

  float mr[2][8], lr[2][8];
#pragma unroll
  for (int t = 0; t < 2; ++t)
#pragma unroll
    for (int r = 0; r < 8; ++r) { mr[t][r] = -3.0e38f; lr[t][r] = 0.0f; }
  v8f o[2][4] = {};

  int nsteps = qb32 / 32 + 1;  // 32-key steps covering keys 0..qb32+31
  for (int kt = 0; kt < nsteps; ++kt) {
    int kb = kt * 32;

    // ---- issue all K and V fragment loads up front (V overlaps softmax) ----
    BF kf[2][2];  // [f][hd-chunk]
#pragma unroll
    for (int f = 0; f < 2; ++f) {
      const __bf16* kr = Kh + (size_t)(kb + f * 16 + ln) * HDdim + hs * 8;
      kf[f][0].h[0] = ld8(kr);      kf[f][0].h[1] = ld8(kr + 16);
      kf[f][1].h[0] = ld8(kr + 32); kf[f][1].h[1] = ld8(kr + 48);
    }
    BF vf[4];
#pragma unroll
    for (int j = 0; j < 4; ++j) {
      const __bf16* vr = Vh + (size_t)(j * 16 + ln) * Sdim + kb + hs * 8;
      vf[j].h[0] = ld8(vr); vf[j].h[1] = ld8(vr + 16);
    }

    // ---- scores + online softmax, both query tiles, both 16-key halves ----
#pragma unroll
    for (int f = 0; f < 2; ++f) {
      int kbf = kb + f * 16;
      v8f s[2];
#pragma unroll
      for (int t = 0; t < 2; ++t) {
        s[t] = v8f{};
        s[t] = wmma_bf16(aq[t][0].v, kf[f][0].v, s[t]);
        s[t] = wmma_bf16(aq[t][1].v, kf[f][1].v, s[t]);
      }
#pragma unroll
      for (int t = 0; t < 2; ++t) {
        int qbt = qb32 + t * 16;
        float sv[8];
#pragma unroll
        for (int r = 0; r < 8; ++r) {
          float v = s[t][r] * 0.125f;  // 1/sqrt(64)
          if (kbf >= qbt) {            // diagonal or fully-masked sub-tile
            int qi = qbt + r + 8 * hs;
            int ki = kbf + ln;
            if (ki > qi) v = -3.0e38f;
          }
          sv[r] = v;
        }
#pragma unroll
        for (int r = 0; r < 8; ++r) {
          float tm = sv[r];
          tm = fmaxf(tm, __shfl_xor(tm, 1));
          tm = fmaxf(tm, __shfl_xor(tm, 2));
          tm = fmaxf(tm, __shfl_xor(tm, 4));
          tm = fmaxf(tm, __shfl_xor(tm, 8));
          float mnew = fmaxf(mr[t][r], tm);
          float corr = __expf(mr[t][r] - mnew);
          float p    = __expf(sv[r] - mnew);
          float rs = p;
          rs += __shfl_xor(rs, 1);
          rs += __shfl_xor(rs, 2);
          rs += __shfl_xor(rs, 4);
          rs += __shfl_xor(rs, 8);
          lr[t][r] = lr[t][r] * corr + rs;
          mr[t][r] = mnew;
#pragma unroll
          for (int j = 0; j < 4; ++j) o[t][j][r] *= corr;
          lds_p[t * 512 + (r + 8 * hs) * 32 + f * 16 + ln] = f2bf(p);
        }
      }
    }

    // ---- reshape P (D-frag in LDS -> A-frag) and accumulate PV ----
    asm volatile("s_wait_dscnt 0" ::: "memory");
#pragma unroll
    for (int t = 0; t < 2; ++t) {
      BF ap;
      const __bf16* pr = lds_p + t * 512 + ln * 32 + hs * 8;
      ap.h[0] = ld8(pr); ap.h[1] = ld8(pr + 16);
#pragma unroll
      for (int j = 0; j < 4; ++j)
        o[t][j] = wmma_bf16(ap.v, vf[j].v, o[t][j]);
    }
  }

  // ---- normalize and store merged-head ctx [B,S,D] bf16 ----
#pragma unroll
  for (int t = 0; t < 2; ++t)
#pragma unroll
    for (int r = 0; r < 8; ++r) {
      float inv = 1.0f / lr[t][r];
      int row = qb32 + t * 16 + r + 8 * hs;
#pragma unroll
      for (int j = 0; j < 4; ++j) {
        ctx[((size_t)b * Sdim + row) * Ddim + h * 64 + j * 16 + ln] = f2bf(o[t][j][r] * inv);
      }
    }
}

// ---------------- kernel 5: output projection (ctx @ W_o + b_o, fp32 out) ----------------
__global__ __launch_bounds__(32) void k_oproj(
    const __bf16* __restrict__ cb, const __bf16* __restrict__ WoT,
    const float* __restrict__ bo, float* __restrict__ out) {
  int lane = threadIdx.x & 31, hs = lane >> 4, ln = lane & 15;
  int mb = blockIdx.x * 32, nb = blockIdx.y * 64;

  v8f acc[2][4] = {};
  mma_32x64(cb, WoT, mb, nb, ln, hs, acc);

#pragma unroll
  for (int j = 0; j < 4; ++j) {
    int n = nb + j * 16 + ln;
    float bvl = bo[n];
#pragma unroll
    for (int i = 0; i < 2; ++i)
#pragma unroll
      for (int r = 0; r < 8; ++r) {
        int m = mb + i * 16 + r + 8 * hs;
        out[(size_t)m * Ddim + n] = acc[i][j][r] + bvl;
      }
  }
}

// ---------------- launcher ----------------
extern "C" void kernel_launch(void* const* d_in, const int* in_sizes, int n_in,
                              void* d_out, int out_size, void* d_ws, size_t ws_size,
                              hipStream_t stream) {
  (void)in_sizes; (void)n_in; (void)out_size; (void)ws_size;
  const float* x  = (const float*)d_in[0];
  const float* Wq = (const float*)d_in[1];
  const float* bq = (const float*)d_in[2];
  const float* Wk = (const float*)d_in[3];
  const float* bk = (const float*)d_in[4];
  const float* Wv = (const float*)d_in[5];
  const float* bv = (const float*)d_in[6];
  const float* Wo = (const float*)d_in[7];
  const float* bo = (const float*)d_in[8];
  float* out = (float*)d_out;

  char* ws = (char*)d_ws;
  const size_t XB = (size_t)Bdim * Sdim * Ddim * 2;   // 8 MiB
  const size_t WT = (size_t)Ddim * Ddim * 2;          // 2 MiB each
  __bf16* xb  = (__bf16*)(ws);
  __bf16* WqT = (__bf16*)(ws + XB);
  __bf16* WkT = (__bf16*)(ws + XB + WT);
  __bf16* WvT = (__bf16*)(ws + XB + 2 * WT);
  __bf16* WoT = (__bf16*)(ws + XB + 3 * WT);
  __bf16* Qb  = (__bf16*)(ws + XB + 4 * WT);
  __bf16* Kb  = (__bf16*)(ws + 2 * XB + 4 * WT);
  __bf16* Vtb = (__bf16*)(ws + 3 * XB + 4 * WT);
  __bf16* ctx = (__bf16*)(ws + 4 * XB + 4 * WT);      // total ~48 MiB

  // 1) x -> bf16
  int n4 = Bdim * Sdim * Ddim / 4;
  k_conv_x<<<dim3(n4 / 256), dim3(256), 0, stream>>>((const float4*)x, (v4bf*)xb, n4);

  // 2) weights -> bf16 transposed
  k_wt<<<dim3(Ddim / 16, Ddim / 16, 4), dim3(16, 16), 0, stream>>>(
      Wq, Wk, Wv, Wo, WqT, WkT, WvT, WoT);

  // 3) QKV projections (32x64 tiles)
  k_qkv<<<dim3((Bdim * Sdim) / 32, Ddim / 64, 3), dim3(32), 0, stream>>>(
      xb, WqT, WkT, WvT, bq, bk, bv, Qb, Kb, Vtb);

  // 4) causal flash attention (32-query bands)
  k_attn<<<dim3(Sdim / 32, Hdim, Bdim), dim3(32), 0, stream>>>(Qb, Kb, Vtb, ctx);

  // 5) output projection (32x64 tiles)
  k_oproj<<<dim3((Bdim * Sdim) / 32, Ddim / 64), dim3(32), 0, stream>>>(ctx, WoT, bo, out);
}
